// MLP_1786706395525
// MI455X (gfx1250) — matrix operand order repaired and checked by reading, not compiled, not run on hardware
//
#include <hip/hip_runtime.h>
#include <hip/hip_bf16.h>
#include <stdint.h>

// ---------------------------------------------------------------------------
// MoE MLP (scattermoe-style) for gfx1250 / MI455X.
//   y_t = sum_{s=0,1} p[t,s] * ( gelu(x_t @ w1[e_ts]) @ w2[e_ts] )
// Strategy: route -> per-expert token lists -> grouped bf16 WMMA GEMMs with
// 2x4 register blocking (8 accumulators per wave iteration).
// ---------------------------------------------------------------------------

typedef __attribute__((ext_vector_type(16))) __bf16       v16bf;
typedef __attribute__((ext_vector_type(8)))  float        v8f;
typedef __attribute__((ext_vector_type(4)))  float        v4f;
typedef __attribute__((ext_vector_type(4)))  __bf16       v4bf;
typedef __attribute__((ext_vector_type(4)))  unsigned int v4u;

union ABFrag {
    v16bf v;
    v4u   q[2];
};

#define N_EXPERTS 8
#define N_TOKENS  4096
#define TOPK      2
#define NPAIRS    (N_TOKENS * TOPK)   // 8192
#define DIN       1024
#define DHID      2048
#define MTILE     32                  // tokens per block
#define TILES_PER_EXPERT ((NPAIRS + MTILE - 1) / MTILE)  // 256 (worst case)

// ---------------------------------------------------------------------------
// Kernel 0: zero the per-expert pair counters.
// ---------------------------------------------------------------------------
__global__ void moe_init_counts(int* __restrict__ counts) {
    if (threadIdx.x < N_EXPERTS) counts[threadIdx.x] = 0;
}

// ---------------------------------------------------------------------------
// Kernel 1: routing.  pair i = token*2 + slot.  Append i to its expert list.
// ---------------------------------------------------------------------------
__global__ void moe_route(const int* __restrict__ expert_idxs,
                          int* __restrict__ counts,
                          int* __restrict__ lists) {
    int i = blockIdx.x * blockDim.x + threadIdx.x;
    if (i < NPAIRS) {
        int e   = expert_idxs[i] & (N_EXPERTS - 1);
        int pos = atomicAdd(&counts[e], 1);
        lists[e * NPAIRS + pos] = i;
    }
}

// ---------------------------------------------------------------------------
// Kernel 2a: w1 [E][DIN][DHID] f32  ->  w1t [E][DHID][DIN] bf16 (transposed)
// so a WMMA B-fragment (one output column, contiguous K) is, per lane, two
// 16-byte runs in memory (k 8h..8h+7 and 16+8h..16+8h+7).
// ---------------------------------------------------------------------------
__global__ void moe_cvt_w1(const float* __restrict__ w1,
                           __bf16* __restrict__ w1t) {
    int i   = blockIdx.x * blockDim.x + threadIdx.x;   // over E*DHID*DIN
    int e   = i >> 21;                                 // / (DHID*DIN)
    int rem = i & ((1 << 21) - 1);
    int h   = rem >> 10;                               // / DIN
    int d   = rem & (DIN - 1);
    w1t[i] = (__bf16)w1[(((size_t)e << 10) + d) * DHID + h];
}

// Kernel 2b: w2 [E][DHID][DIN] f32 -> w2t [E][DIN][DHID] bf16 (transposed)
__global__ void moe_cvt_w2(const float* __restrict__ w2,
                           __bf16* __restrict__ w2t) {
    int i   = blockIdx.x * blockDim.x + threadIdx.x;   // over E*DIN*DHID
    int e   = i >> 21;
    int rem = i & ((1 << 21) - 1);
    int d   = rem >> 11;                               // / DHID
    int h   = rem & (DHID - 1);
    w2t[i] = (__bf16)w2[(((size_t)e << 11) + h) * DIN + d];
}

// ---------------------------------------------------------------------------
// Branch-free tanh-approximate gelu (jax.nn.gelu default).  Keeps EXEC
// uniform and avoids the branchy libm tanhf expansion.
// ---------------------------------------------------------------------------
__device__ __forceinline__ float gelu_tanh(float v) {
    float u = 0.7978845608028654f * (v + 0.044715f * v * v * v);
#if __has_builtin(__builtin_amdgcn_tanhf)
    return 0.5f * v * (1.0f + __builtin_amdgcn_tanhf(u));   // v_tanh_f32
#else
    // tanh(u) = 1 - 2/(e^{2u}+1)  ->  gelu = v * (1 - rcp(e^{2u}+1))
    float E = __expf(2.0f * u);
    return v * (1.0f - __builtin_amdgcn_rcpf(E + 1.0f));
#endif
}

// ---------------------------------------------------------------------------
// Kernel 3: fused grouped MLP.  One block = (expert e, tile of 32 pairs).
// 256 threads = 8 waves (wave32).  Dynamic LDS (192.25 KB):
//   Xs [32][1024] bf16  (64 KB)   staged input rows
//   Hs [32][2048] bf16  (128 KB)  gelu(x@w1)*gate, bf16
//   sh_gate[32] f32, sh_pair[32] int
// Each wave iteration register-blocks 2 M-tiles x 4 N-tiles (8 WMMA accs).
// ---------------------------------------------------------------------------
__global__ __launch_bounds__(256) void moe_mlp_wmma(
        const float*  __restrict__ x,        // [N_TOKENS][DIN]
        const float*  __restrict__ gates,    // [N_TOKENS][TOPK] flat
        const int*    __restrict__ counts,   // [E]
        const int*    __restrict__ lists,    // [E][NPAIRS]
        const __bf16* __restrict__ w1t,      // [E][DHID][DIN]
        const __bf16* __restrict__ w2t,      // [E][DIN][DHID]
        float*        __restrict__ partial)  // [NPAIRS][DIN]
{
    extern __shared__ char smem[];
    __bf16* Xs      = (__bf16*)smem;                              // 65536 B
    __bf16* Hs      = (__bf16*)(smem + MTILE * DIN * 2);          // 131072 B
    float*  sh_gate = (float*)(smem + MTILE * DIN * 2 + MTILE * DHID * 2);
    int*    sh_pair = (int*)(sh_gate + MTILE);

    const int e     = blockIdx.x >> 8;          // expert
    const int tilei = blockIdx.x & 255;         // tile within expert
    const int base  = tilei * MTILE;
    const int cnt   = counts[e];
    if (base >= cnt) return;                    // block-uniform exit

    const int tid = threadIdx.x;

    // ---- stage X tile (f32 -> bf16) + gates + pair ids -------------------
    {
        int r     = tid >> 3;           // row 0..31, 8 threads per row
        int cpart = (tid & 7) * 128;    // 128 floats per thread
        int gidx  = base + r;
        int pair  = (gidx < cnt) ? lists[e * NPAIRS + gidx] : -1;
        if ((tid & 7) == 0) {
            sh_pair[r] = pair;
            sh_gate[r] = (pair >= 0) ? gates[pair] : 0.0f;
        }
        __bf16* dst = Xs + r * DIN + cpart;
        if (pair >= 0) {
            const float* src = x + ((size_t)(pair >> 1)) * DIN + cpart;
            #pragma unroll 4
            for (int c = 0; c < 128; c += 4) {
                v4f f = *(const v4f*)(src + c);
                v4bf b;
                b[0] = (__bf16)f[0]; b[1] = (__bf16)f[1];
                b[2] = (__bf16)f[2]; b[3] = (__bf16)f[3];
                *(v4bf*)(dst + c) = b;
            }
        } else {
            v4bf z = {};
            #pragma unroll 4
            for (int c = 0; c < 128; c += 4) *(v4bf*)(dst + c) = z;
        }
    }
    __syncthreads();

    const int  wv   = tid >> 5;        // wave 0..7
    const int  lane = tid & 31;
    const int  lr   = lane & 15;
    const bool hi   = lane >= 16;
    const int  akb  = hi ? 8 : 0;      // A-fragment K base within 32
    const int  bkb  = hi ? 8 : 0;      // B-fragment K base within 32
    const int  mofs = hi ? 8 : 0;      // C/D row offset for this lane half

    // ---- GEMM1: H[32][2048] = gelu(X @ W1[e]) * gate, bf16 in LDS --------
    // N-groups of 4 tiles; each group computes 2 (M) x 4 (N) accumulators.
    for (int g = wv; g < DHID / 64; g += 8) {             // 32 groups
        const int tn0 = g * 4;
        v8f acc[2][4] = {};
        const __bf16* arow0 = Xs + lr * DIN + akb;        // M-tile 0
        const __bf16* arow1 = Xs + (16 + lr) * DIN + akb; // M-tile 1
        const __bf16* bcol  = w1t + (((size_t)e * DHID) + tn0 * 16 + lr) * DIN + bkb;
        __builtin_prefetch((const void*)bcol, 0, 1);
        #pragma unroll 2
        for (int kb = 0; kb < DIN / 32; ++kb) {
            ABFrag a0, a1, b[4];
            a0.q[0] = *(const v4u*)(arow0 + kb * 32);
            a0.q[1] = *(const v4u*)(arow0 + kb * 32 + 16);
            a1.q[0] = *(const v4u*)(arow1 + kb * 32);
            a1.q[1] = *(const v4u*)(arow1 + kb * 32 + 16);
            #pragma unroll
            for (int j = 0; j < 4; ++j) {
                b[j].q[0] = *(const v4u*)(bcol + (size_t)j * 16 * DIN + kb * 32);
                b[j].q[1] = *(const v4u*)(bcol + (size_t)j * 16 * DIN + kb * 32 + 16);
            }
            #pragma unroll
            for (int j = 0; j < 4; ++j) {
                acc[0][j] = __builtin_amdgcn_wmma_f32_16x16x32_bf16(
                        false, a0.v, false, b[j].v, (short)0, acc[0][j], false, false);
                acc[1][j] = __builtin_amdgcn_wmma_f32_16x16x32_bf16(
                        false, a1.v, false, b[j].v, (short)0, acc[1][j], false, false);
            }
        }
        #pragma unroll
        for (int tm = 0; tm < 2; ++tm) {
            #pragma unroll
            for (int j = 0; j < 4; ++j) {
                #pragma unroll
                for (int r = 0; r < 8; ++r) {
                    int   m = tm * 16 + r + mofs;
                    float gt = sh_gate[m];
                    Hs[m * DHID + (tn0 + j) * 16 + lr] =
                        (__bf16)(gt * gelu_tanh(acc[tm][j][r]));
                }
            }
        }
    }
    __syncthreads();

    // ---- GEMM2: Y[32][1024] = H @ W2[e] -> partial[pair][:] --------------
    for (int g = wv; g < DIN / 64; g += 8) {              // 16 groups
        const int tn0 = g * 4;
        v8f acc[2][4] = {};
        const __bf16* arow0 = Hs + lr * DHID + akb;
        const __bf16* arow1 = Hs + (16 + lr) * DHID + akb;
        const __bf16* bcol  = w2t + (((size_t)e * DIN) + tn0 * 16 + lr) * DHID + bkb;
        __builtin_prefetch((const void*)bcol, 0, 1);
        #pragma unroll 2
        for (int kb = 0; kb < DHID / 32; ++kb) {
            ABFrag a0, a1, b[4];
            a0.q[0] = *(const v4u*)(arow0 + kb * 32);
            a0.q[1] = *(const v4u*)(arow0 + kb * 32 + 16);
            a1.q[0] = *(const v4u*)(arow1 + kb * 32);
            a1.q[1] = *(const v4u*)(arow1 + kb * 32 + 16);
            #pragma unroll
            for (int j = 0; j < 4; ++j) {
                b[j].q[0] = *(const v4u*)(bcol + (size_t)j * 16 * DHID + kb * 32);
                b[j].q[1] = *(const v4u*)(bcol + (size_t)j * 16 * DHID + kb * 32 + 16);
            }
            #pragma unroll
            for (int j = 0; j < 4; ++j) {
                acc[0][j] = __builtin_amdgcn_wmma_f32_16x16x32_bf16(
                        false, a0.v, false, b[j].v, (short)0, acc[0][j], false, false);
                acc[1][j] = __builtin_amdgcn_wmma_f32_16x16x32_bf16(
                        false, a1.v, false, b[j].v, (short)0, acc[1][j], false, false);
            }
        }
        #pragma unroll
        for (int tm = 0; tm < 2; ++tm) {
            #pragma unroll
            for (int j = 0; j < 4; ++j) {
                #pragma unroll
                for (int r = 0; r < 8; ++r) {
                    int m    = tm * 16 + r + mofs;
                    int pair = sh_pair[m];
                    if (pair >= 0)
                        partial[(size_t)pair * DIN + (tn0 + j) * 16 + lr] =
                            acc[tm][j][r];
                }
            }
        }
    }
}

// ---------------------------------------------------------------------------
// Kernel 4: combine the two top-k slot partials into the final output.
// ---------------------------------------------------------------------------
__global__ void moe_combine(const float* __restrict__ partial,
                            float* __restrict__ out) {
    int i = blockIdx.x * blockDim.x + threadIdx.x;   // over N_TOKENS*DIN
    int t = i >> 10;
    int c = i & (DIN - 1);
    size_t p0 = ((size_t)t * 2) * DIN + c;
    out[i] = partial[p0] + partial[p0 + DIN];
}

// ---------------------------------------------------------------------------
// Host-side launcher (graph-capture safe: only kernel launches on `stream`).
// ---------------------------------------------------------------------------
extern "C" void kernel_launch(void* const* d_in, const int* in_sizes, int n_in,
                              void* d_out, int out_size, void* d_ws, size_t ws_size,
                              hipStream_t stream) {
    const float* x        = (const float*)d_in[0];   // [2,2048,1024]
    const float* expert_p = (const float*)d_in[1];   // [4096,2]
    const int*   eidx     = (const int*)d_in[2];     // [4096,2]
    const float* w1       = (const float*)d_in[3];   // [8,1024,2048]
    const float* w2       = (const float*)d_in[4];   // [8,2048,1024]
    float*       out      = (float*)d_out;           // [2,2048,1024]

    // Workspace layout (bytes):
    char*   ws      = (char*)d_ws;
    int*    counts  = (int*)ws;                              // 8 ints (pad 256)
    int*    lists   = (int*)(ws + 256);                      // 8*8192 ints
    size_t  off     = 256 + (size_t)N_EXPERTS * NPAIRS * sizeof(int);
    __bf16* w1t     = (__bf16*)(ws + off);                   // 33.55 MB
    off            += (size_t)N_EXPERTS * DHID * DIN * sizeof(__bf16);
    __bf16* w2t     = (__bf16*)(ws + off);                   // 33.55 MB
    off            += (size_t)N_EXPERTS * DIN * DHID * sizeof(__bf16);
    float*  partial = (float*)(ws + off);                    // 8192*1024 f32

    moe_init_counts<<<1, 32, 0, stream>>>(counts);
    moe_route<<<NPAIRS / 256, 256, 0, stream>>>(eidx, counts, lists);

    int wlelems = N_EXPERTS * DHID * DIN;                    // 16,777,216
    moe_cvt_w1<<<wlelems / 256, 256, 0, stream>>>(w1, w1t);
    moe_cvt_w2<<<wlelems / 256, 256, 0, stream>>>(w2, w2t);

    size_t smem = (size_t)MTILE * DIN * 2 + (size_t)MTILE * DHID * 2
                + MTILE * sizeof(float) + MTILE * sizeof(int);  // 196,864 B
    moe_mlp_wmma<<<N_EXPERTS * TILES_PER_EXPERT, 256, smem, stream>>>(
        x, expert_p, counts, lists, w1t, w2t, partial);

    moe_combine<<<(N_TOKENS * DIN) / 256, 256, 0, stream>>>(partial, out);
}
